// TABL_55662776156282
// MI455X (gfx1250) — matrix operand order repaired and checked by reading, not compiled, not run on hardware
//
#include <hip/hip_runtime.h>
#include <hip/hip_bf16.h>

typedef __attribute__((ext_vector_type(16))) _Float16 v16h;
typedef __attribute__((ext_vector_type(8)))  float    v8f;

#define BN      65536
#define DD      40
#define TT      10
#define HH      120
#define T2      5
#define CC      3
#define NB      8          // batch elements per chunk
#define JC      80         // NB*TT columns per chunk
#define NCHUNK  8192       // BN/NB
#define G1      256        // workgroups for the GEMM passes
#define NSLOT1  1200       // H*T softmax slots
#define NSLOT2  15         // C*T2 softmax slots
#define KP      72         // padded K stride (halfs) for f16 LDS images
#define I1S     84         // it1 row stride (floats), 16B-aligned, bank-skewed

// two contiguous 16B LDS segments -> one v16h fragment (no branches)
__device__ __forceinline__ v16h ld2x16(const _Float16* p0, const _Float16* p1) {
    union { uint4 u[2]; v16h h; } cv;
    cv.u[0] = *(const uint4*)p0;
    cv.u[1] = *(const uint4*)p1;
    return cv.h;
}

// stage W1_1 as zero-padded f16 image [128][KP]; A frag = 4 aligned 16B loads
__device__ __forceinline__ void stage_wa(_Float16* wa, const float* __restrict__ W1_1,
                                         int tid) {
    for (int i = tid; i < 128 * 64; i += 256) {
        const int h = i >> 6, k = i & 63;
        wa[h * KP + k] = (h < HH && k < DD) ? (_Float16)W1_1[h * DD + k] : (_Float16)0.f;
    }
}

// ---------------------------------------------------------------------------
// K1: WMMA item1 = W1_1 @ x, z = item1 @ W_1, online softmax stats over batch
// ---------------------------------------------------------------------------
__global__ __launch_bounds__(256) void k1_stats(const float* __restrict__ x,
                                                const float* __restrict__ W1_1,
                                                const float* __restrict__ W_1,
                                                float* __restrict__ pm1,
                                                float* __restrict__ ps1) {
    __shared__ alignas(16) _Float16 wa[128 * KP];    // W1_1 f16, zero-padded
    __shared__ alignas(16) _Float16 xs[JC * KP];     // x chunk, column-major [j][k]
    __shared__ alignas(16) float    it1[128 * I1S];  // item1 [h][j]
    __shared__ float w1s[TT * TT];

    const int tid  = threadIdx.x;
    const int lane = tid & 31;
    const int wave = tid >> 5;             // 0..7 == H tile
    const int hi   = (lane >= 16) ? 1 : 0;
    const int nrow = lane & 15;

    if (tid < TT * TT) w1s[tid] = W_1[tid];
    stage_wa(wa, W1_1, tid);
    // zero K-padding rows of xs ONCE (never overwritten by chunk staging)
    for (int i = tid; i < JC * (64 - DD); i += 256) {
        const int j = i / (64 - DD), k = DD + i % (64 - DD);
        xs[j * KP + k] = (_Float16)0.f;
    }
    __syncthreads();

    // A fragments: loop-invariant vector loads
    const _Float16* war = wa + (wave * 16 + nrow) * KP;
    const v16h a0 = ld2x16(war + hi * 8,      war + 16 + hi * 8);
    const v16h a1 = ld2x16(war + 32 + hi * 8, war + 48 + hi * 8);

    float om[5], os[5];
#pragma unroll
    for (int i = 0; i < 5; ++i) { om[i] = -__builtin_inff(); os[i] = 0.f; }

    for (int chunk = blockIdx.x; chunk < NCHUNK; chunk += gridDim.x) {
        const size_t bbase = (size_t)chunk * NB;
        __syncthreads();
        for (int i = tid; i < NB * DD * TT; i += 256) {
            const int lb = i / (DD * TT), rem = i % (DD * TT);
            const int d = rem / TT, t = rem % TT;
            xs[(lb * TT + t) * KP + d] = (_Float16)x[(bbase + lb) * (DD * TT) + rem];
        }
        if (chunk + gridDim.x < NCHUNK)
            __builtin_prefetch(x + (bbase + (size_t)gridDim.x * NB) * (DD * TT) + tid * 12, 0, 0);
        __syncthreads();

        // each wave: 5 column tiles of 16; K=40 padded to 64 -> 2 WMMAs
        for (int ct = 0; ct < 5; ++ct) {
            const int jb = ct * 16;
            const _Float16* xc = xs + (jb + nrow) * KP;
            const v16h b0 = ld2x16(xc + hi * 8,      xc + 16 + hi * 8);
            const v16h b1 = ld2x16(xc + 32 + hi * 8, xc + 48 + hi * 8);
            v8f acc = {};
            acc = __builtin_amdgcn_wmma_f32_16x16x32_f16(false, a0, false, b0,
                                                         (short)0, acc, false, false);
            acc = __builtin_amdgcn_wmma_f32_16x16x32_f16(false, a1, false, b1,
                                                         (short)0, acc, false, false);
            float* outp = it1 + (wave * 16 + hi * 8) * I1S + jb + nrow;
#pragma unroll
            for (int r = 0; r < 8; ++r) outp[r * I1S] = acc[r];   // unconditional
        }
        __syncthreads();

        // z = item1 @ W_1 ; online (max, sum-exp) per owned slot
#pragma unroll
        for (int i = 0; i < 5; ++i) {
            const int s = tid + i * 256;
            if (i < 4 || tid < NSLOT1 - 1024) {
                const int h = s / TT, t = s % TT;
                for (int lb = 0; lb < NB; ++lb) {
                    float z = 0.f;
#pragma unroll
                    for (int tp = 0; tp < TT; ++tp)
                        z += it1[h * I1S + lb * TT + tp] * w1s[tp * TT + t];
                    if (z > om[i]) { os[i] = os[i] * __expf(om[i] - z) + 1.f; om[i] = z; }
                    else            { os[i] += __expf(z - om[i]); }
                }
            }
        }
    }
#pragma unroll
    for (int i = 0; i < 5; ++i) {
        const int s = tid + i * 256;
        if (i < 4 || tid < NSLOT1 - 1024) {
            pm1[blockIdx.x * NSLOT1 + s] = om[i];
            ps1[blockIdx.x * NSLOT1 + s] = os[i];
        }
    }
}

// ---------------------------------------------------------------------------
// K2: merge per-workgroup online-softmax partials -> global max / sum-exp
// ---------------------------------------------------------------------------
__global__ void k_reduce(const float* __restrict__ pm, const float* __restrict__ ps,
                         float* __restrict__ M, float* __restrict__ S,
                         int nslots, int nwg) {
    const int s = blockIdx.x * blockDim.x + threadIdx.x;
    if (s >= nslots) return;
    float m = -__builtin_inff();
    for (int w = 0; w < nwg; ++w) m = fmaxf(m, pm[w * nslots + s]);
    float sum = 0.f;
    for (int w = 0; w < nwg; ++w) sum += ps[w * nslots + s] * __expf(pm[w * nslots + s] - m);
    M[s] = m; S[s] = sum;
}

// ---------------------------------------------------------------------------
// K3: recompute item1/z, blend -> item4 (in place), item5 (relu), item6 ->
//     workspace, online stats for second softmax
// ---------------------------------------------------------------------------
__global__ __launch_bounds__(256) void k3_main(const float* __restrict__ x,
                                               const float* __restrict__ W1_1,
                                               const float* __restrict__ W_1,
                                               const float* __restrict__ alpha_p,
                                               const float* __restrict__ W2_1,
                                               const float* __restrict__ bias_1,
                                               const float* __restrict__ W1_2,
                                               const float* __restrict__ W_2,
                                               const float* __restrict__ M1,
                                               const float* __restrict__ S1,
                                               float* __restrict__ it6g,
                                               float* __restrict__ pm2,
                                               float* __restrict__ ps2) {
    __shared__ alignas(16) _Float16 wa[128 * KP];
    __shared__ alignas(16) _Float16 xs[JC * KP];
    __shared__ alignas(16) float    it1[128 * I1S];
    __shared__ float it5[HH * 41];
    __shared__ float it6s[CC * NB * T2];
    __shared__ float w1s[100], w21s[50], b1ls[600], w12s[360], w2s[25];

    const int tid  = threadIdx.x;
    const int lane = tid & 31;
    const int wave = tid >> 5;
    const int hi   = (lane >= 16) ? 1 : 0;
    const int nrow = lane & 15;

    if (tid < 100) w1s[tid]  = W_1[tid];
    if (tid < 50)  w21s[tid] = W2_1[tid];
    if (tid < 25)  w2s[tid]  = W_2[tid];
    for (int i = tid; i < 600; i += 256) b1ls[i] = bias_1[i];
    for (int i = tid; i < 360; i += 256) w12s[i] = W1_2[i];
    stage_wa(wa, W1_1, tid);
    for (int i = tid; i < JC * (64 - DD); i += 256) {
        const int j = i / (64 - DD), k = DD + i % (64 - DD);
        xs[j * KP + k] = (_Float16)0.f;
    }
    __syncthreads();

    const float alpha = alpha_p[0];
    const float onem  = 1.f - alpha;

    const _Float16* war = wa + (wave * 16 + nrow) * KP;
    const v16h a0 = ld2x16(war + hi * 8,      war + 16 + hi * 8);
    const v16h a1 = ld2x16(war + 32 + hi * 8, war + 48 + hi * 8);

    float m1r[5], is1r[5];
#pragma unroll
    for (int i = 0; i < 5; ++i) {
        const int s = tid + i * 256;
        const bool act = (i < 4) || (tid < NSLOT1 - 1024);
        m1r[i]  = act ? M1[s] : 0.f;
        is1r[i] = act ? 1.f / S1[s] : 0.f;
    }
    float om2 = -__builtin_inff(), os2 = 0.f;

    for (int chunk = blockIdx.x; chunk < NCHUNK; chunk += gridDim.x) {
        const size_t bbase = (size_t)chunk * NB;
        __syncthreads();
        for (int i = tid; i < NB * DD * TT; i += 256) {
            const int lb = i / (DD * TT), rem = i % (DD * TT);
            const int d = rem / TT, t = rem % TT;
            xs[(lb * TT + t) * KP + d] = (_Float16)x[(bbase + lb) * (DD * TT) + rem];
        }
        if (chunk + gridDim.x < NCHUNK)
            __builtin_prefetch(x + (bbase + (size_t)gridDim.x * NB) * (DD * TT) + tid * 12, 0, 0);
        __syncthreads();

        for (int ct = 0; ct < 5; ++ct) {
            const int jb = ct * 16;
            const _Float16* xc = xs + (jb + nrow) * KP;
            const v16h b0 = ld2x16(xc + hi * 8,      xc + 16 + hi * 8);
            const v16h b1 = ld2x16(xc + 32 + hi * 8, xc + 48 + hi * 8);
            v8f acc = {};
            acc = __builtin_amdgcn_wmma_f32_16x16x32_f16(false, a0, false, b0,
                                                         (short)0, acc, false, false);
            acc = __builtin_amdgcn_wmma_f32_16x16x32_f16(false, a1, false, b1,
                                                         (short)0, acc, false, false);
            float* outp = it1 + (wave * 16 + hi * 8) * I1S + jb + nrow;
#pragma unroll
            for (int r = 0; r < 8; ++r) outp[r * I1S] = acc[r];
        }
        __syncthreads();

        // z for all owned slots (item1 still needed by every t' reader)
        float zb[5][NB];
#pragma unroll
        for (int i = 0; i < 5; ++i) {
            const int s = tid + i * 256;
            if (i < 4 || tid < NSLOT1 - 1024) {
                const int h = s / TT, t = s % TT;
                for (int lb = 0; lb < NB; ++lb) {
                    float z = 0.f;
#pragma unroll
                    for (int tp = 0; tp < TT; ++tp)
                        z += it1[h * I1S + lb * TT + tp] * w1s[tp * TT + t];
                    zb[i][lb] = z;
                }
            }
        }
        __syncthreads();
        // item4 = item1 * (alpha*softmax + (1-alpha)), in place
#pragma unroll
        for (int i = 0; i < 5; ++i) {
            const int s = tid + i * 256;
            if (i < 4 || tid < NSLOT1 - 1024) {
                const int h = s / TT, t = s % TT;
                for (int lb = 0; lb < NB; ++lb) {
                    const float f = alpha * __expf(zb[i][lb] - m1r[i]) * is1r[i] + onem;
                    it1[h * I1S + lb * TT + t] *= f;
                }
            }
        }
        __syncthreads();

        // item5 = relu(item4 @ W2_1 + bias_1)   (120*8*5 outputs)
        for (int idx = tid; idx < HH * NB * T2; idx += 256) {
            const int h = idx / (NB * T2), rem = idx % (NB * T2);
            const int lb = rem / T2, u = rem % T2;
            float v = b1ls[h * T2 + u];
#pragma unroll
            for (int tp = 0; tp < TT; ++tp)
                v += it1[h * I1S + lb * TT + tp] * w21s[tp * T2 + u];
            it5[h * 41 + lb * T2 + u] = fmaxf(v, 0.f);
        }
        __syncthreads();

        // item6 = W1_2 @ item5  (3*8*5 outputs, 120-deep)
        if (tid < CC * NB * T2) {
            const int c = tid / (NB * T2), rem = tid % (NB * T2);
            const int lb = rem / T2, u = rem % T2;
            float v = 0.f;
            for (int h = 0; h < HH; ++h)
                v += w12s[c * HH + h] * it5[h * 41 + lb * T2 + u];
            it6s[c * (NB * T2) + rem] = v;
            it6g[(bbase + lb) * (CC * T2) + c * T2 + u] = v;
        }
        __syncthreads();

        // z2 = item6 @ W_2 ; online stats over batch (15 slots)
        if (tid < NSLOT2) {
            const int c = tid / T2, u = tid % T2;
            for (int lb = 0; lb < NB; ++lb) {
                float z = 0.f;
#pragma unroll
                for (int up = 0; up < T2; ++up)
                    z += it6s[c * (NB * T2) + lb * T2 + up] * w2s[up * T2 + u];
                if (z > om2) { os2 = os2 * __expf(om2 - z) + 1.f; om2 = z; }
                else          { os2 += __expf(z - om2); }
            }
        }
    }
    if (tid < NSLOT2) {
        pm2[blockIdx.x * NSLOT2 + tid] = om2;
        ps2[blockIdx.x * NSLOT2 + tid] = os2;
    }
}

// ---------------------------------------------------------------------------
// K5: finish from cached item6 (L2-resident, 3.9 MB)
// ---------------------------------------------------------------------------
__global__ __launch_bounds__(256) void k5_final(const float* __restrict__ it6g,
                                                const float* __restrict__ W_2,
                                                const float* __restrict__ W2_2,
                                                const float* __restrict__ bias_2,
                                                const float* __restrict__ alpha_p,
                                                const float* __restrict__ M2,
                                                const float* __restrict__ S2,
                                                float* __restrict__ out) {
    const int b = blockIdx.x * blockDim.x + threadIdx.x;
    if (b >= BN) return;
    const float alpha = alpha_p[0];
    const float onem  = 1.f - alpha;
    float t6[CC * T2];
#pragma unroll
    for (int i = 0; i < CC * T2; ++i) t6[i] = it6g[(size_t)b * (CC * T2) + i];
#pragma unroll
    for (int c = 0; c < CC; ++c) {
        float o = bias_2[c];
#pragma unroll
        for (int u = 0; u < T2; ++u) {
            float z2 = 0.f;
#pragma unroll
            for (int up = 0; up < T2; ++up)
                z2 += t6[c * T2 + up] * W_2[up * T2 + u];
            const int s = c * T2 + u;
            const float f = alpha * __expf(z2 - M2[s]) / S2[s] + onem;
            o += t6[c * T2 + u] * f * W2_2[u];
        }
        out[b * CC + c] = o;
    }
}

// ---------------------------------------------------------------------------
extern "C" void kernel_launch(void* const* d_in, const int* in_sizes, int n_in,
                              void* d_out, int out_size, void* d_ws, size_t ws_size,
                              hipStream_t stream) {
    const float* x      = (const float*)d_in[0];
    const float* W1_1   = (const float*)d_in[1];
    const float* W_1    = (const float*)d_in[2];
    const float* alpha  = (const float*)d_in[3];
    const float* W2_1   = (const float*)d_in[4];
    const float* bias_1 = (const float*)d_in[5];
    const float* W1_2   = (const float*)d_in[6];
    const float* W_2    = (const float*)d_in[7];
    const float* W2_2   = (const float*)d_in[8];
    const float* bias_2 = (const float*)d_in[9];
    float* out = (float*)d_out;

    float* ws  = (float*)d_ws;
    float* pm1 = ws;                              // G1*1200
    float* ps1 = pm1 + (size_t)G1 * NSLOT1;       // G1*1200
    float* M1  = ps1 + (size_t)G1 * NSLOT1;       // 1200
    float* S1  = M1 + NSLOT1;                     // 1200
    float* pm2 = S1 + NSLOT1;                     // G1*15
    float* ps2 = pm2 + (size_t)G1 * NSLOT2;       // G1*15
    float* M2  = ps2 + (size_t)G1 * NSLOT2;       // 15
    float* S2  = M2 + NSLOT2;                     // 15
    float* it6 = S2 + NSLOT2;                     // BN*15

    k1_stats<<<G1, 256, 0, stream>>>(x, W1_1, W_1, pm1, ps1);
    k_reduce<<<(NSLOT1 + 255) / 256, 256, 0, stream>>>(pm1, ps1, M1, S1, NSLOT1, G1);
    k3_main<<<G1, 256, 0, stream>>>(x, W1_1, W_1, alpha, W2_1, bias_1, W1_2, W_2,
                                    M1, S1, it6, pm2, ps2);
    k_reduce<<<1, 256, 0, stream>>>(pm2, ps2, M2, S2, NSLOT2, G1);
    k5_final<<<BN / 256, 256, 0, stream>>>(it6, W_2, W2_2, bias_2, alpha, M2, S2, out);
}
